// CIGA_21560735825997
// MI455X (gfx1250) — matrix operand order repaired and checked by reading, not compiled, not run on hardware
//
#include <hip/hip_runtime.h>
#include <hip/hip_bf16.h>

#define N_NODES  20000
#define N_GRAPHS 200
#define NPG      100
#define N_EDGES  100000
#define D_IN     64
#define EMB      300
#define NL       5
#define N_OUT    10
#define EPG      500
#define KTOP     125
#define NCAUSAL  (N_GRAPHS * KTOP)   // 25000

typedef __bf16 bf16;
typedef bf16  bf16x8 __attribute__((ext_vector_type(8)));
typedef bf16  v16bf  __attribute__((ext_vector_type(16)));
typedef float v8f    __attribute__((ext_vector_type(8)));

__device__ __forceinline__ bf16 f2bf(float f) {
  unsigned u = __float_as_uint(f);
  unsigned r = u + 0x7FFFu + ((u >> 16) & 1u);   // round-to-nearest-even
  unsigned short h = (unsigned short)(r >> 16);
  bf16 o; __builtin_memcpy(&o, &h, 2);
  return o;
}

// ---------------- bf16 WMMA GEMM ----------------
// Requires: K % 4 == 0, N % 4 == 0, KH % 4 == 0 (true for all call sites:
// K in {64,300,600,1200 subsets}, N in {300,600,1200}, KH=300). Rows of all
// f32 operands are 16B aligned (K*4 and N*4 multiples of 16).
#define BM 128
#define BN 128
#define BK 64
#define LDT 72   // padded LDS K-stride (bf16): 144B rows -> conflict-free b128 frag loads

// A fragment, 16x32 bf16 (ISA 7.12.2): lanes 0-15 row M=lane, K {0..7,16..23};
// lanes 16-31 row M=lane-16, K {8..15,24..31}. Two 16B LDS loads.
__device__ __forceinline__ v16bf load_fragA(const bf16* row, int laneHi) {
  bf16x8 lo = *reinterpret_cast<const bf16x8*>(row + 8 * laneHi);
  bf16x8 hi = *reinterpret_cast<const bf16x8*>(row + 16 + 8 * laneHi);
  v16bf r;
#pragma unroll
  for (int i = 0; i < 8; ++i) { r[i] = lo[i]; r[i + 8] = hi[i]; }
  return r;
}
// B fragment, 32x16 bf16: lane = 16*kh + n holds K {8*kh+(0..7), 16+8*kh+(0..7)}, column n.
// B tile stored transposed in LDS as Bt[n][k] -> two 16B loads.
__device__ __forceinline__ v16bf load_fragB(const bf16* row, int laneHi) {
  bf16x8 lo = *reinterpret_cast<const bf16x8*>(row + 8 * laneHi);
  bf16x8 hi = *reinterpret_cast<const bf16x8*>(row + 16 + 8 * laneHi);
  v16bf r;
#pragma unroll
  for (int i = 0; i < 8; ++i) { r[i] = lo[i]; r[i + 8] = hi[i]; }
  return r;
}

// AMODE 0: A is a plain [M x K] f32 matrix.
// AMODE 1: A row m = [H[gsrc[m]][0:KH] || H[gdst[m]][0:KH]]  (edge_rep gather, K = 2*KH)
// FUSE_DOT: instead of storing C, compute sum_n relu(C[m][n]+bias[n])*w2[n] -> atomicAdd score[m]
template <int AMODE, bool RELU, bool FUSE_DOT>
__global__ __launch_bounds__(256) void gemm_bf16_kernel(
    const float* __restrict__ A, const float* __restrict__ B,
    const float* __restrict__ bias, float* __restrict__ C,
    int M, int N, int K,
    const int* __restrict__ gsrc, const int* __restrict__ gdst,
    const float* __restrict__ H, int KH,
    const float* __restrict__ w2, float* __restrict__ score)
{
  __shared__ bf16 As[BM * LDT];
  __shared__ bf16 Bs[BN * LDT];   // transposed: row = n (0..127), col = k (0..63)

  const int tid    = threadIdx.x;
  const int lane   = tid & 31;
  const int laneLo = lane & 15;
  const int laneHi = lane >> 4;
  const int wave   = tid >> 5;    // 0..7
  const int waveM  = wave & 1;    // 2 wave rows -> 128 rows
  const int waveN  = wave >> 1;   // 4 wave cols -> 128 cols
  const int blockM = blockIdx.x * BM;
  const int blockN = blockIdx.y * BN;

  v8f acc[4][2];
#pragma unroll
  for (int a = 0; a < 4; ++a)
#pragma unroll
    for (int b = 0; b < 2; ++b)
#pragma unroll
      for (int r = 0; r < 8; ++r) acc[a][b][r] = 0.0f;

  for (int kb = 0; kb < K; kb += BK) {
    // ---- load A tile (BM x BK), f32 -> bf16; quad-granular guards only ----
    {
      const int col  = (tid & 15) * 4;  // 0..60
      const int row0 = tid >> 4;        // 0..15
#pragma unroll
      for (int p = 0; p < 8; ++p) {
        const int row = row0 + p * 16;
        const int gm  = blockM + row;
        const int gk  = kb + col;
        float4 q = make_float4(0.f, 0.f, 0.f, 0.f);
        if (AMODE == 0) {
          if (gm < M && gk < K)
            q = *reinterpret_cast<const float4*>(A + (size_t)gm * K + gk);
        } else {
          if (gm < M && gk < K) {
            // KH % 4 == 0 -> a quad never straddles the src/dst boundary
            if (gk < KH)
              q = *reinterpret_cast<const float4*>(H + (size_t)gsrc[gm] * KH + gk);
            else
              q = *reinterpret_cast<const float4*>(H + (size_t)gdst[gm] * KH + (gk - KH));
          }
        }
        bf16* dp = &As[row * LDT + col];
        dp[0] = f2bf(q.x); dp[1] = f2bf(q.y); dp[2] = f2bf(q.z); dp[3] = f2bf(q.w);
      }
    }
    // ---- load B tile (BK x BN) transposed into Bt[n][k] ----
    {
      const int n4 = (tid & 31) * 4;    // 0..124
      const int k0 = tid >> 5;          // 0..7
#pragma unroll
      for (int p = 0; p < 8; ++p) {
        const int k  = k0 + p * 8;
        const int gk = kb + k;
        const int gn = blockN + n4;
        float4 q = make_float4(0.f, 0.f, 0.f, 0.f);
        if (gk < K && gn < N)   // N % 4 == 0 -> quad fully in or out
          q = *reinterpret_cast<const float4*>(B + (size_t)gk * N + gn);
        bf16* bp = &Bs[n4 * LDT + k];
        bp[0 * LDT] = f2bf(q.x); bp[1 * LDT] = f2bf(q.y);
        bp[2 * LDT] = f2bf(q.z); bp[3 * LDT] = f2bf(q.w);
      }
    }
    __syncthreads();

#pragma unroll
    for (int kk = 0; kk < BK; kk += 32) {
      v16bf af[4];
#pragma unroll
      for (int mf = 0; mf < 4; ++mf)
        af[mf] = load_fragA(&As[(waveM * 64 + mf * 16 + laneLo) * LDT + kk], laneHi);
      v16bf bfr[2];
#pragma unroll
      for (int nf = 0; nf < 2; ++nf)
        bfr[nf] = load_fragB(&Bs[(waveN * 32 + nf * 16 + laneLo) * LDT + kk], laneHi);
#pragma unroll
      for (int mf = 0; mf < 4; ++mf)
#pragma unroll
        for (int nf = 0; nf < 2; ++nf)
          acc[mf][nf] = __builtin_amdgcn_wmma_f32_16x16x32_bf16(
              false, af[mf], false, bfr[nf], (short)0, acc[mf][nf], false, false);
    }
    __syncthreads();
  }

  if (!FUSE_DOT) {
#pragma unroll
    for (int mf = 0; mf < 4; ++mf)
#pragma unroll
      for (int nf = 0; nf < 2; ++nf) {
        const int n = blockN + waveN * 32 + nf * 16 + laneLo;
        if (n < N) {
          const float bs = bias ? bias[n] : 0.f;
#pragma unroll
          for (int r = 0; r < 8; ++r) {
            const int m = blockM + waveM * 64 + mf * 16 + r + 8 * laneHi;
            if (m < M) {
              float v = acc[mf][nf][r] + bs;
              if (RELU) v = fmaxf(v, 0.f);
              C[(size_t)m * N + n] = v;
            }
          }
        }
      }
  } else {
#pragma unroll
    for (int mf = 0; mf < 4; ++mf) {
      float part[8];
#pragma unroll
      for (int r = 0; r < 8; ++r) part[r] = 0.f;
#pragma unroll
      for (int nf = 0; nf < 2; ++nf) {
        const int n = blockN + waveN * 32 + nf * 16 + laneLo;
        const float bs = (n < N) ? bias[n] : 0.f;
        const float wn = (n < N) ? w2[n]   : 0.f;
#pragma unroll
        for (int r = 0; r < 8; ++r)
          part[r] += fmaxf(acc[mf][nf][r] + bs, 0.f) * wn;
      }
#pragma unroll
      for (int r = 0; r < 8; ++r) {
        float p = part[r];
        p += __shfl_xor(p, 1);
        p += __shfl_xor(p, 2);
        p += __shfl_xor(p, 4);
        p += __shfl_xor(p, 8);
        if (laneLo == 0) {
          const int m = blockM + waveM * 64 + mf * 16 + r + 8 * laneHi;
          if (m < M) atomicAdd(&score[m], p);
        }
      }
    }
  }
}

// ---------------- helper kernels ----------------
__global__ void fill_kernel(float* p, float v, long long n) {
  long long i = (long long)blockIdx.x * blockDim.x + threadIdx.x;
  const long long st = (long long)gridDim.x * blockDim.x;
  for (; i < n; i += st) p[i] = v;
}

__global__ void fill_from_kernel(float* p, const float* v, long long n) {
  long long i = (long long)blockIdx.x * blockDim.x + threadIdx.x;
  const long long st = (long long)gridDim.x * blockDim.x;
  const float vv = v[0];
  for (; i < n; i += st) p[i] = vv;
}

// block-per-edge scatter: agg[dst[e]][d] += h[src[e]][d] * (ew ? ew[e] : 1)
template <bool HAS_W>
__global__ void seg_sum_kernel(const float* __restrict__ h, const int* __restrict__ src,
                               const int* __restrict__ dst, const float* __restrict__ ew,
                               float* __restrict__ agg, int E, int D) {
  const int e = blockIdx.x;
  if (e >= E) return;
  const size_t sbase = (size_t)src[e] * D;
  const size_t dbase = (size_t)dst[e] * D;
  const float w = HAS_W ? ew[e] : 1.0f;
  for (int d = threadIdx.x; d < D; d += blockDim.x) {
    float v = h[sbase + d];
    if (HAS_W) v *= w;
    atomicAdd(&agg[dbase + d], v);
  }
}

// agg = (1+eps[l])*h + agg  (in place over agg)
__global__ void axpy_eps_kernel(const float* __restrict__ h, float* __restrict__ agg,
                                const float* __restrict__ eps, int l, long long n) {
  const float c = 1.0f + eps[l];
  long long i = (long long)blockIdx.x * blockDim.x + threadIdx.x;
  const long long st = (long long)gridDim.x * blockDim.x;
  for (; i < n; i += st) agg[i] = c * h[i] + agg[i];
}

// per-graph top-K of edge scores (500 edges/graph, bitonic over 512 in LDS)
__global__ __launch_bounds__(256) void topk_kernel(
    const float* __restrict__ score, const int* __restrict__ src, const int* __restrict__ dst,
    int* __restrict__ c_src, int* __restrict__ c_dst, float* __restrict__ c_w,
    float* __restrict__ mask) {
  __shared__ float sk[512];
  __shared__ int   si[512];
  const int g = blockIdx.x, t = threadIdx.x;
  for (int i = t; i < 512; i += 256) {
    sk[i] = (i < EPG) ? score[g * EPG + i] : -3.0e38f;
    si[i] = i;
  }
  __syncthreads();
  for (int k = 2; k <= 512; k <<= 1) {
    for (int j = k >> 1; j > 0; j >>= 1) {
      for (int i = t; i < 512; i += 256) {
        const int ixj = i ^ j;
        if (ixj > i) {
          const bool desc = ((i & k) == 0);
          const float a = sk[i], b = sk[ixj];
          if (desc ? (a < b) : (a > b)) {
            sk[i] = b; sk[ixj] = a;
            const int ti = si[i]; si[i] = si[ixj]; si[ixj] = ti;
          }
        }
      }
      __syncthreads();
    }
  }
  for (int i = t; i < KTOP; i += 256) {
    const int e = g * EPG + si[i];
    const int s = src[e], d = dst[e];
    c_src[g * KTOP + i] = s;
    c_dst[g * KTOP + i] = d;
    c_w[g * KTOP + i]   = 1.f / (1.f + expf(-sk[i]));
    mask[s] = 1.f;   // benign races: all write 1.0
    mask[d] = 1.f;
  }
}

// masked mean pool per graph (nodes g*100 .. g*100+99)
__global__ void pool_kernel(const float* __restrict__ hc, const float* __restrict__ mask,
                            float* __restrict__ pooled) {
  const int g = blockIdx.x;
  __shared__ float s_inv;
  if (threadIdx.x == 0) {
    float c = 0.f;
    for (int n = 0; n < NPG; ++n) c += mask[g * NPG + n];
    s_inv = 1.0f / fmaxf(c, 1.0f);
  }
  __syncthreads();
  const float inv = s_inv;
  for (int d = threadIdx.x; d < EMB; d += blockDim.x) {
    float s = 0.f;
    for (int n = 0; n < NPG; ++n) {
      const int node = g * NPG + n;
      s += hc[(size_t)node * EMB + d] * mask[node];
    }
    pooled[g * EMB + d] = s * inv;
  }
}

__global__ void pred_kernel(const float* __restrict__ pooled, const float* __restrict__ W,
                            const float* __restrict__ b, float* __restrict__ out) {
  const int i = blockIdx.x * blockDim.x + threadIdx.x;
  if (i >= N_GRAPHS * N_OUT) return;
  const int g = i / N_OUT, o = i % N_OUT;
  float s = b[o];
  for (int d = 0; d < EMB; ++d) s += pooled[g * EMB + d] * W[d * N_OUT + o];
  out[i] = s;
}

static inline int ceildiv(int a, int b) { return (a + b - 1) / b; }

extern "C" void kernel_launch(void* const* d_in, const int* in_sizes, int n_in,
                              void* d_out, int out_size, void* d_ws, size_t ws_size,
                              hipStream_t stream) {
  (void)in_sizes; (void)n_in; (void)out_size; (void)ws_size;
  const float* x        = (const float*)d_in[0];
  const int*   eidx     = (const int*)d_in[1];
  const int*   src      = eidx;
  const int*   dst      = eidx + N_EDGES;
  const float* enc_in_W = (const float*)d_in[4];
  const float* enc_in_b = (const float*)d_in[5];
  const float* enc_W1   = (const float*)d_in[6];
  const float* enc_b1   = (const float*)d_in[7];
  const float* enc_W2   = (const float*)d_in[8];
  const float* enc_b2   = (const float*)d_in[9];
  const float* enc_eps  = (const float*)d_in[10];
  const float* att_W1   = (const float*)d_in[11];
  const float* att_b1   = (const float*)d_in[12];
  const float* att_W2   = (const float*)d_in[13];
  const float* att_b2   = (const float*)d_in[14];
  const float* clf_in_W = (const float*)d_in[15];
  const float* clf_in_b = (const float*)d_in[16];
  const float* clf_W1   = (const float*)d_in[17];
  const float* clf_b1   = (const float*)d_in[18];
  const float* clf_W2   = (const float*)d_in[19];
  const float* clf_b2   = (const float*)d_in[20];
  const float* clf_eps  = (const float*)d_in[21];
  const float* pred_W   = (const float*)d_in[22];
  const float* pred_b   = (const float*)d_in[23];

  float* ws     = (float*)d_ws;
  float* hA     = ws;                  // 6.0M f32
  float* hB     = ws + 6000000;        // 6.0M
  float* agg    = ws + 12000000;       // 6.0M
  float* tbuf   = ws + 18000000;       // 12.0M (20000 x 600)
  float* score  = ws + 30000000;       // 100k
  int*   c_src  = (int*)(ws + 30100000);
  int*   c_dst  = (int*)(ws + 30125000);
  float* c_w    = ws + 30150000;
  float* mask   = ws + 30175000;
  float* pooled = ws + 30195000;       // 60k

  const dim3 blk(256);
  const long long nhe = (long long)N_NODES * EMB;

  // ---------- encoder GIN ----------
  gemm_bf16_kernel<0, false, false><<<dim3(ceildiv(N_NODES, BM), ceildiv(EMB, BN)), blk, 0, stream>>>(
      x, enc_in_W, enc_in_b, hA, N_NODES, EMB, D_IN,
      nullptr, nullptr, nullptr, 0, nullptr, nullptr);

  float* hc = hA; float* ho = hB;
  for (int l = 0; l < NL; ++l) {
    fill_kernel<<<2048, 256, 0, stream>>>(agg, 0.f, nhe);
    seg_sum_kernel<false><<<N_EDGES, 256, 0, stream>>>(hc, src, dst, nullptr, agg, N_EDGES, EMB);
    axpy_eps_kernel<<<2048, 256, 0, stream>>>(hc, agg, enc_eps, l, nhe);
    gemm_bf16_kernel<0, true, false><<<dim3(ceildiv(N_NODES, BM), ceildiv(2 * EMB, BN)), blk, 0, stream>>>(
        agg, enc_W1 + (size_t)l * EMB * 2 * EMB, enc_b1 + l * 2 * EMB, tbuf,
        N_NODES, 2 * EMB, EMB, nullptr, nullptr, nullptr, 0, nullptr, nullptr);
    if (l < NL - 1)
      gemm_bf16_kernel<0, true, false><<<dim3(ceildiv(N_NODES, BM), ceildiv(EMB, BN)), blk, 0, stream>>>(
          tbuf, enc_W2 + (size_t)l * 2 * EMB * EMB, enc_b2 + l * EMB, ho,
          N_NODES, EMB, 2 * EMB, nullptr, nullptr, nullptr, 0, nullptr, nullptr);
    else
      gemm_bf16_kernel<0, false, false><<<dim3(ceildiv(N_NODES, BM), ceildiv(EMB, BN)), blk, 0, stream>>>(
          tbuf, enc_W2 + (size_t)l * 2 * EMB * EMB, enc_b2 + l * EMB, ho,
          N_NODES, EMB, 2 * EMB, nullptr, nullptr, nullptr, 0, nullptr, nullptr);
    float* tmp = hc; hc = ho; ho = tmp;
  }

  // ---------- edge scores: score[e] = relu([h_s||h_d]@W1 + b1)@W2 + b2 (fused) ----------
  fill_from_kernel<<<ceildiv(N_EDGES, 256), 256, 0, stream>>>(score, att_b2, (long long)N_EDGES);
  gemm_bf16_kernel<1, false, true><<<dim3(ceildiv(N_EDGES, BM), ceildiv(4 * EMB, BN)), blk, 0, stream>>>(
      nullptr, att_W1, att_b1, nullptr, N_EDGES, 4 * EMB, 2 * EMB,
      src, dst, hc, EMB, att_W2, score);

  // ---------- per-graph top-K + node mask ----------
  fill_kernel<<<ceildiv(N_NODES, 256), 256, 0, stream>>>(mask, 0.f, (long long)N_NODES);
  topk_kernel<<<N_GRAPHS, 256, 0, stream>>>(score, src, dst, c_src, c_dst, c_w, mask);

  // ---------- classifier GIN on causal subgraph ----------
  gemm_bf16_kernel<0, false, false><<<dim3(ceildiv(N_NODES, BM), ceildiv(EMB, BN)), blk, 0, stream>>>(
      x, clf_in_W, clf_in_b, ho, N_NODES, EMB, D_IN,
      nullptr, nullptr, nullptr, 0, nullptr, nullptr);
  float* kc = ho; float* ko = hc;
  for (int l = 0; l < NL; ++l) {
    fill_kernel<<<2048, 256, 0, stream>>>(agg, 0.f, nhe);
    seg_sum_kernel<true><<<NCAUSAL, 256, 0, stream>>>(kc, c_src, c_dst, c_w, agg, NCAUSAL, EMB);
    axpy_eps_kernel<<<2048, 256, 0, stream>>>(kc, agg, clf_eps, l, nhe);
    gemm_bf16_kernel<0, true, false><<<dim3(ceildiv(N_NODES, BM), ceildiv(2 * EMB, BN)), blk, 0, stream>>>(
        agg, clf_W1 + (size_t)l * EMB * 2 * EMB, clf_b1 + l * 2 * EMB, tbuf,
        N_NODES, 2 * EMB, EMB, nullptr, nullptr, nullptr, 0, nullptr, nullptr);
    if (l < NL - 1)
      gemm_bf16_kernel<0, true, false><<<dim3(ceildiv(N_NODES, BM), ceildiv(EMB, BN)), blk, 0, stream>>>(
          tbuf, clf_W2 + (size_t)l * 2 * EMB * EMB, clf_b2 + l * EMB, ko,
          N_NODES, EMB, 2 * EMB, nullptr, nullptr, nullptr, 0, nullptr, nullptr);
    else
      gemm_bf16_kernel<0, false, false><<<dim3(ceildiv(N_NODES, BM), ceildiv(EMB, BN)), blk, 0, stream>>>(
          tbuf, clf_W2 + (size_t)l * 2 * EMB * EMB, clf_b2 + l * EMB, ko,
          N_NODES, EMB, 2 * EMB, nullptr, nullptr, nullptr, 0, nullptr, nullptr);
    float* tmp = kc; kc = ko; ko = tmp;
  }

  // ---------- pooling + prediction ----------
  pool_kernel<<<N_GRAPHS, 256, 0, stream>>>(kc, mask, pooled);
  pred_kernel<<<ceildiv(N_GRAPHS * N_OUT, 256), 256, 0, stream>>>(pooled, pred_W, pred_b, (float*)d_out);
}
